// HAN_91061896610464
// MI455X (gfx1250) — matrix operand order repaired and checked by reading, not compiled, not run on hardware
//
#include <hip/hip_runtime.h>
#include <hip/hip_bf16.h>
#include <math.h>

// ---------------------------------------------------------------------------
// CDNA5 (gfx1250) HAN: bf16 WMMA GEMMs + WMMA-driven LSTM scans + TDM staging.
// wave32; 256-thread blocks = 8 waves. All K dims forced to multiples of 32
// (x is zero-padded 52->64) so fragment loads are straight-line b128 loads.
// ---------------------------------------------------------------------------

typedef __attribute__((ext_vector_type(16))) __bf16 v16bf;
typedef __attribute__((ext_vector_type(8)))  float  v8f;

#if defined(__has_builtin)
#if __has_builtin(__builtin_amdgcn_tensor_load_to_lds)
#define USE_TDM 1
#endif
#endif
#ifndef USE_TDM
#define USE_TDM 0
#endif

#if USE_TDM
typedef __attribute__((ext_vector_type(4))) unsigned int v4u;
typedef __attribute__((ext_vector_type(8))) int          v8i;
typedef __attribute__((ext_vector_type(4))) int          v4i;
#endif

#define N_NOTES 8192
#define N_BEATS 1024
#define N_MEASQ 256
#define IN_DIM  52
#define IN_PAD  64
#define FINAL_H 128

__device__ __forceinline__ float sigf(float x) { return 1.f / (1.f + __expf(-x)); }

// ---------------------------------------------------------------------------
// Zero-pad x[8192,52] -> xpad[8192,64] so the note-level GEMM K is 32-aligned.
// ---------------------------------------------------------------------------
__global__ void k_padx(const float* __restrict__ x, float* __restrict__ o, int total) {
    int i = blockIdx.x * blockDim.x + threadIdx.x;
    if (i >= total) return;
    int r = i >> 6, c = i & 63;
    o[i] = (c < IN_DIM) ? x[r * IN_DIM + c] : 0.f;
}

// ---------------------------------------------------------------------------
// Pack W[N, Krow] (f32, row-major, columns [koff, koff+Kreal)) into bf16
// WMMA-B fragments: fragment (nt, kt) = 512 bf16 at (nt*Ktiles + kt)*512;
// (lane l, elem e): col = nt*16 + (l&15), k = kt*32 + (l>=16 ? 16:0) + e.
// K padded with zeros to Ktiles*32.
// ---------------------------------------------------------------------------
__global__ void k_packB(const float* __restrict__ W, int N, int Krow, int koff,
                        int Kreal, int Ktiles, __bf16* __restrict__ out, int total) {
    int id = blockIdx.x * blockDim.x + threadIdx.x;
    if (id >= total) return;
    int r  = id & 511;
    int f  = id >> 9;
    int kt = f % Ktiles;
    int nt = f / Ktiles;
    int l  = r >> 4;
    int e  = r & 15;
    int n  = nt * 16 + (l & 15);
    int k  = kt * 32 + ((l >> 4) << 4) + e;
    float v = (k < Kreal) ? W[(size_t)n * Krow + koff + k] : 0.f;
    out[id] = (__bf16)v;
}

// b = x + y (combined bias bi+bh)
__global__ void k_addvec(const float* __restrict__ x, const float* __restrict__ y,
                         float* __restrict__ o, int n) {
    int i = blockIdx.x * blockDim.x + threadIdx.x;
    if (i < n) o[i] = x[i] + y[i];
}

// Build A fragment (contiguous k order, as in k_packB) from two contiguous 8-float runs:
// elems 0..7  -> K = base + khalf + 0..7
// elems 8..15 -> K = base + khalf + 8..15       (khalf = 0 or 16 by half-wave)
__device__ __forceinline__ v16bf load_a_frag(const float* __restrict__ Arow,
                                             int base, int khalf) {
    const float4* p0 = (const float4*)(Arow + base + khalf);
    const float4* p1 = (const float4*)(Arow + base + khalf + 8);
    float4 f0 = p0[0], f1 = p0[1], f2 = p1[0], f3 = p1[1];
    v16bf a;
    a[0] = (__bf16)f0.x;  a[1] = (__bf16)f0.y;  a[2] = (__bf16)f0.z;  a[3] = (__bf16)f0.w;
    a[4] = (__bf16)f1.x;  a[5] = (__bf16)f1.y;  a[6] = (__bf16)f1.z;  a[7] = (__bf16)f1.w;
    a[8] = (__bf16)f2.x;  a[9] = (__bf16)f2.y;  a[10] = (__bf16)f2.z; a[11] = (__bf16)f2.w;
    a[12] = (__bf16)f3.x; a[13] = (__bf16)f3.y; a[14] = (__bf16)f3.z; a[15] = (__bf16)f3.w;
    return a;
}

// ---------------------------------------------------------------------------
// C[M,N] (f32) = gatherA(A)[M,K] * packedB + bias  (or += when accumulate).
// Block: 256 thr = 8 waves; tile 128(M) x 64(N); wave w owns a 16-row strip.
// B fragments staged to LDS by the Tensor Data Mover (wave 0 issues a 2D D#:
// tile 512 x 4 bf16, row stride Ktiles*512) with s_wait_tensorcnt before the
// barrier; cooperative-copy fallback when the TDM builtin is unavailable.
// K must be a multiple of 32 (host guarantees via padding).
// ---------------------------------------------------------------------------
__global__ void k_gemm(const float* __restrict__ A, int lda,
                       const int* __restrict__ rowidx,
                       const __bf16* __restrict__ Bp,
                       const float* __restrict__ bias,
                       float* __restrict__ C, int ldc,
                       int Ktiles, int N, int accumulate) {
    __shared__ __bf16 bsh[4 * 512];   // 4 n-tiles * 512 bf16 = 4 KB
    const int w = threadIdx.x >> 5;
    const int l = threadIdx.x & 31;
    const int mbase = blockIdx.y * 128 + w * 16;
    const int nbase = blockIdx.x * 64;
    const int nt0   = nbase >> 4;

    v8f acc[4] = {v8f{0}, v8f{0}, v8f{0}, v8f{0}};

    const int row  = mbase + (l & 15);
    const int srow = rowidx ? rowidx[row] : row;
    const float* Arow = A + (size_t)srow * lda;
    const int khalf = (l >> 4) * 16;

    for (int kt = 0; kt < Ktiles; ++kt) {
        if (kt + 1 < Ktiles)   // warm L2->L0 for next slab (global_prefetch_b8)
            __builtin_prefetch(Bp + (((size_t)nt0 * Ktiles + kt + 1) << 9) + l * 16, 0, 1);
#if USE_TDM
        if (threadIdx.x < 32) {   // wave 0 drives the Tensor Data Mover
            unsigned long long ga =
                (unsigned long long)(Bp + (((size_t)nt0 * Ktiles + kt) << 9));
            unsigned int lds = (unsigned int)(size_t)&bsh[0];
            unsigned int td0 = (unsigned int)(Ktiles << 9);  // elems per n-tile row
            unsigned int td1 = (unsigned int)(N >> 4);       // total n-tiles
            unsigned long long st0 = (unsigned long long)(Ktiles << 9);
            v4u g0;
            g0[0] = 1u;                                   // count=1, user D#
            g0[1] = lds;                                  // lds_addr (bytes)
            g0[2] = (unsigned int)ga;                     // global_addr[31:0]
            g0[3] = ((unsigned int)(ga >> 32) & 0x1FFFFFFu) | (2u << 30); // [56:32], type=2
            v8i g1;
            g1[0] = (int)(1u << 16);                      // wg_mask=0, data_size=2B
            g1[1] = (int)((td0 & 0xFFFFu) << 16);         // tensor_dim0[15:0]
            g1[2] = (int)((td0 >> 16) | ((td1 & 0xFFFFu) << 16));
            g1[3] = (int)((td1 >> 16) | (512u << 16));    // tile_dim0 = 512
            g1[4] = (int)4;                               // tile_dim1 = 4, tile_dim2 = 0
            g1[5] = (int)(st0 & 0xFFFFFFFFu);             // tensor_dim0_stride lo
            g1[6] = (int)((st0 >> 32) & 0xFFFFu);         // stride hi | dim1_stride lo
            g1[7] = 0;
            v4i z4 = {0, 0, 0, 0};
#if defined(__clang_major__) && __clang_major__ >= 23
            v8i z8 = {0, 0, 0, 0, 0, 0, 0, 0};
            __builtin_amdgcn_tensor_load_to_lds(g0, g1, z4, z4, z8, 0);
#else
            __builtin_amdgcn_tensor_load_to_lds(g0, g1, z4, z4, 0);
#endif
            __builtin_amdgcn_s_wait_tensorcnt(0);
        }
        __syncthreads();
#else
        {
            int idx = threadIdx.x * 8;                 // 2048 bf16 total
            int f   = idx >> 9, r = idx & 511;
            const __bf16* src = Bp + (((size_t)(nt0 + f) * Ktiles + kt) << 9) + r;
            *(uint4*)&bsh[idx] = *(const uint4*)src;   // 16B = 8 bf16
        }
        __syncthreads();
#endif
        v16bf a = load_a_frag(Arow, kt * 32, khalf);
#pragma unroll
        for (int ct = 0; ct < 4; ++ct) {
            v16bf b = *(const v16bf*)&bsh[ct * 512 + l * 16];
            acc[ct] = __builtin_amdgcn_wmma_f32_16x16x32_bf16(
                false, a, false, b, (short)0, acc[ct], false, false);
        }
        __syncthreads();
    }

    // C/D layout: vgpr v, lanes0-15 -> (M=v, N=lane); lanes16-31 -> (M=v+8)
#pragma unroll
    for (int ct = 0; ct < 4; ++ct) {
#pragma unroll
        for (int v = 0; v < 8; ++v) {
            int r = mbase + v + ((l >> 4) << 3);
            int c = nbase + ct * 16 + (l & 15);
            float val = acc[ct][v] + (bias ? bias[c] : 0.f);
            size_t off = (size_t)r * ldc + c;
            if (accumulate) C[off] += val; else C[off] = val;
        }
    }
}

// ---------------------------------------------------------------------------
// LSTM scan. grid.x = 1 (uni) or 2 (fwd+bwd). 256 threads = 8 waves.
// xg[t] = x@Wi.T + bi + bh (precomputed). Per step, h@Wh.T via WMMA with h
// broadcast into A row 0 (lanes 0/16): the fragment is two aligned 16B LDS
// loads + a row-0 cndmask; D row 0 (acc[0], lanes 0-15) is the gate slice.
// ---------------------------------------------------------------------------
__global__ void k_lstm_scan(const float* __restrict__ xgF, const float* __restrict__ xgB,
                            const __bf16* __restrict__ WhF, const __bf16* __restrict__ WhB,
                            float* __restrict__ hout, int ldh,
                            float* __restrict__ hlast,
                            const float* __restrict__ h0, const float* __restrict__ c0,
                            int S, int H) {
    const int dir = blockIdx.x;
    const float*  xg  = dir ? xgB : xgF;
    const __bf16* Whp = dir ? WhB : WhF;
    const int hoff = dir ? H : 0;
    const int H4 = 4 * H;
    const int Ktiles = H >> 5;
    const int tpw = (H4 >> 4) >> 3;   // n-tiles per wave (8 waves)

    __shared__ __bf16 hsh[256];
    __shared__ float  gsh[1024];

    const int tid = threadIdx.x, w = tid >> 5, l = tid & 31;
    float c = 0.f, h = 0.f;
    if (tid < H) {
        h = h0 ? h0[tid] : 0.f;
        c = c0 ? c0[tid] : 0.f;
        hsh[tid] = (__bf16)h;
    }
    __syncthreads();

    const int  khalf  = (l >> 4) * 16;
    const bool lane_lo = (l < 16);
    const bool row0   = ((l & 15) == 0);   // only A row 0 carries h
    const uint4 zz = make_uint4(0u, 0u, 0u, 0u);

    for (int t = 0; t < S; ++t) {
        const int te = dir ? (S - 1 - t) : t;
        const float* xgt = xg + (size_t)te * H4;

        for (int tt = 0; tt < tpw; ++tt) {
            const int nt = w * tpw + tt;
            const __bf16* Wtile = Whp + (((size_t)nt * Ktiles) << 9) + l * 16;
            v8f acc = v8f{0};
            for (int kt = 0; kt < Ktiles; ++kt) {
                uint4 lo = *(const uint4*)(hsh + kt * 32 + khalf);
                uint4 hi = *(const uint4*)(hsh + kt * 32 + khalf + 8);
                union { uint4 u[2]; v16bf v; } au;
                au.u[0] = row0 ? lo : zz;
                au.u[1] = row0 ? hi : zz;
                v16bf b = *(const v16bf*)(Wtile + ((size_t)kt << 9));
                acc = __builtin_amdgcn_wmma_f32_16x16x32_bf16(
                    false, au.v, false, b, (short)0, acc, false, false);
            }
            if (lane_lo) {                       // D row 0: vgpr 0, lanes 0-15
                int col = nt * 16 + l;
                gsh[col] = acc[0] + xgt[col];
            }
        }
        __syncthreads();

        if (tid < H) {                           // gate order: i, f, g, o
            float gi = gsh[tid], gf = gsh[H + tid];
            float gg = gsh[2 * H + tid], go = gsh[3 * H + tid];
            c = sigf(gf) * c + sigf(gi) * tanhf(gg);
            h = sigf(go) * tanhf(c);
            hsh[tid] = (__bf16)h;
            if (hout) hout[(size_t)te * ldh + hoff + tid] = h;
        }
        __syncthreads();
    }
    if (hlast && dir == 0 && tid < H) hlast[tid] = h;
}

// ---------------------------------------------------------------------------
// Per-segment, per-feature softmax-weighted sum (segments sorted+contiguous).
// ---------------------------------------------------------------------------
__global__ void k_segatt(const float* __restrict__ h, const float* __restrict__ a,
                         const int* __restrict__ segid, int S, int D,
                         float* __restrict__ out) {
    const int s = blockIdx.x;
    __shared__ int lohi[2];
    if (threadIdx.x == 0) {
        int lo = 0, hi = S;
        while (lo < hi) { int m = (lo + hi) >> 1; if (segid[m] < s) lo = m + 1; else hi = m; }
        lohi[0] = lo;
        int lo2 = lo, hi2 = S;
        while (lo2 < hi2) { int m = (lo2 + hi2) >> 1; if (segid[m] < s + 1) lo2 = m + 1; else hi2 = m; }
        lohi[1] = lo2;
    }
    __syncthreads();
    const int lo = lohi[0], hi = lohi[1];
    for (int d = threadIdx.x; d < D; d += blockDim.x) {
        float mx = -3.402823466e38f;
        for (int t = lo; t < hi; ++t) mx = fmaxf(mx, a[(size_t)t * D + d]);
        float se = 0.f, sh = 0.f;
        for (int t = lo; t < hi; ++t) {
            float e = __expf(a[(size_t)t * D + d] - mx);
            se += e;
            sh += e * h[(size_t)t * D + d];
        }
        out[(size_t)s * D + d] = sh / se;
    }
}

// beat_measure[b] = measure of the beat's first note (segment_min equivalent).
__global__ void k_first_measure(const int* __restrict__ beat, const int* __restrict__ meas,
                                int S, int* __restrict__ beatmeas) {
    int i = blockIdx.x * blockDim.x + threadIdx.x;
    if (i >= S) return;
    if (i == 0 || beat[i] != beat[i - 1]) beatmeas[beat[i]] = meas[i];
}

// Final MLP head: relu(fc1([h_last, years])) @ fc2 + b
__global__ void k_head(const float* __restrict__ hlast, const float* __restrict__ years,
                       const float* __restrict__ fc1W, const float* __restrict__ fc1b,
                       const float* __restrict__ fc2W, const float* __restrict__ fc2b,
                       float* __restrict__ out) {
    __shared__ float s1[32];
    int j = threadIdx.x;
    if (j < 32) {
        float acc = fc1b[j];
        const float* wr = fc1W + j * (FINAL_H + 9);
        for (int k = 0; k < FINAL_H; ++k) acc += wr[k] * hlast[k];
        for (int k = 0; k < 9; ++k)       acc += wr[FINAL_H + k] * years[k];
        s1[j] = fmaxf(acc, 0.f);
    }
    __syncthreads();
    if (j == 0) {
        float o = fc2b[0];
        for (int k = 0; k < 32; ++k) o += fc2W[k] * s1[k];
        out[0] = o;
    }
}

// ---------------------------------------------------------------------------
// Host launcher
// ---------------------------------------------------------------------------
enum {
    I_X = 0, I_YEARS = 1, I_FH0 = 2, I_FC0 = 3,
    I_NOTE = 4,            // 2 layers x {Wi_f,Wh_f,bi_f,bh_f,Wi_b,Wh_b,bi_b,bh_b}
    I_BEAT = 20,           // 2 layers x 8
    I_MEAS = 36,           // 1 layer  x 8
    I_OUT_WI = 44, I_OUT_WH = 45, I_OUT_BI = 46, I_OUT_BH = 47,
    I_BATT_W = 48, I_BATT_B = 49, I_MATT_W = 50, I_MATT_B = 51,
    I_FC1W = 52, I_FC1B = 53, I_FC2W = 54, I_FC2B = 55,
    I_BEATNUM = 56, I_MEASNUM = 57
};

extern "C" void kernel_launch(void* const* d_in, const int* in_sizes, int n_in,
                              void* d_out, int out_size, void* d_ws, size_t ws_size,
                              hipStream_t stream) {
    (void)in_sizes; (void)n_in; (void)out_size; (void)ws_size;
    const float* x       = (const float*)d_in[I_X];
    const float* years   = (const float*)d_in[I_YEARS];
    const float* fh0     = (const float*)d_in[I_FH0];
    const float* fc0     = (const float*)d_in[I_FC0];
    const int*   beatnum = (const int*)d_in[I_BEATNUM];
    const int*   measnum = (const int*)d_in[I_MEASNUM];

    char* ws = (char*)d_ws;
    size_t off = 0;
    auto alloc = [&](size_t bytes) -> void* {
        void* p = ws + off;
        off = (off + bytes + 255) & ~(size_t)255;
        return p;
    };

    auto pack = [&](int idx, int N, int Krow, int koff, int Kreal) -> __bf16* {
        int Ktiles = (Kreal + 31) / 32;
        int total  = (N / 16) * Ktiles * 512;
        __bf16* out = (__bf16*)alloc((size_t)total * sizeof(__bf16));
        k_packB<<<dim3((total + 255) / 256), dim3(256), 0, stream>>>(
            (const float*)d_in[idx], N, Krow, koff, Kreal, Ktiles, out, total);
        return out;
    };
    auto bias_sum = [&](int ia, int ib, int n) -> float* {
        float* b = (float*)alloc((size_t)n * sizeof(float));
        k_addvec<<<dim3((n + 255) / 256), dim3(256), 0, stream>>>(
            (const float*)d_in[ia], (const float*)d_in[ib], b, n);
        return b;
    };
    auto gemm = [&](float* C, int ldc, const float* A, int lda, const int* rowidx,
                    const __bf16* Bp, const float* bias, int M, int Kpad, int N, int accum) {
        k_gemm<<<dim3(N / 64, M / 128), dim3(256), 0, stream>>>(
            A, lda, rowidx, Bp, bias, C, ldc, Kpad / 32, N, accum);
    };

    // ---- packed weights (bf16 fragment order) --------------------------------
    __bf16* pWi_n0f = pack(I_NOTE + 0, 1024, IN_DIM, 0, IN_DIM);  // Ktiles=2 (pad to 64)
    __bf16* pWh_n0f = pack(I_NOTE + 1, 1024, 256, 0, 256);
    __bf16* pWi_n0b = pack(I_NOTE + 4, 1024, IN_DIM, 0, IN_DIM);
    __bf16* pWh_n0b = pack(I_NOTE + 5, 1024, 256, 0, 256);
    __bf16* pWi_n1f = pack(I_NOTE + 8, 1024, 512, 0, 512);
    __bf16* pWh_n1f = pack(I_NOTE + 9, 1024, 256, 0, 256);
    __bf16* pWi_n1b = pack(I_NOTE + 12, 1024, 512, 0, 512);
    __bf16* pWh_n1b = pack(I_NOTE + 13, 1024, 256, 0, 256);
    __bf16* pWi_b0f = pack(I_BEAT + 0, 1024, 512, 0, 512);
    __bf16* pWh_b0f = pack(I_BEAT + 1, 1024, 256, 0, 256);
    __bf16* pWi_b0b = pack(I_BEAT + 4, 1024, 512, 0, 512);
    __bf16* pWh_b0b = pack(I_BEAT + 5, 1024, 256, 0, 256);
    __bf16* pWi_b1f = pack(I_BEAT + 8, 1024, 512, 0, 512);
    __bf16* pWh_b1f = pack(I_BEAT + 9, 1024, 256, 0, 256);
    __bf16* pWi_b1b = pack(I_BEAT + 12, 1024, 512, 0, 512);
    __bf16* pWh_b1b = pack(I_BEAT + 13, 1024, 256, 0, 256);
    __bf16* pWi_mf = pack(I_MEAS + 0, 512, 512, 0, 512);
    __bf16* pWh_mf = pack(I_MEAS + 1, 512, 128, 0, 128);
    __bf16* pWi_mb = pack(I_MEAS + 4, 512, 512, 0, 512);
    __bf16* pWh_mb = pack(I_MEAS + 5, 512, 128, 0, 128);
    // out LSTM Wi[512,1280] split by K-range: 'combined' never materialized
    __bf16* pWiO_a = pack(I_OUT_WI, 512, 1280, 0, 512);     // h_note part
    __bf16* pWiO_b = pack(I_OUT_WI, 512, 1280, 512, 512);   // h_beat gather part
    __bf16* pWiO_c = pack(I_OUT_WI, 512, 1280, 1024, 256);  // h_meas gather part
    __bf16* pWh_o  = pack(I_OUT_WH, 512, 128, 0, 128);
    __bf16* pAttB = pack(I_BATT_W, 512, 512, 0, 512);
    __bf16* pAttM = pack(I_MATT_W, 512, 512, 0, 512);

    // ---- combined biases (bi + bh) -------------------------------------------
    float* b_n0f = bias_sum(I_NOTE + 2, I_NOTE + 3, 1024);
    float* b_n0b = bias_sum(I_NOTE + 6, I_NOTE + 7, 1024);
    float* b_n1f = bias_sum(I_NOTE + 10, I_NOTE + 11, 1024);
    float* b_n1b = bias_sum(I_NOTE + 14, I_NOTE + 15, 1024);
    float* b_b0f = bias_sum(I_BEAT + 2, I_BEAT + 3, 1024);
    float* b_b0b = bias_sum(I_BEAT + 6, I_BEAT + 7, 1024);
    float* b_b1f = bias_sum(I_BEAT + 10, I_BEAT + 11, 1024);
    float* b_b1b = bias_sum(I_BEAT + 14, I_BEAT + 15, 1024);
    float* b_mf  = bias_sum(I_MEAS + 2, I_MEAS + 3, 512);
    float* b_mb  = bias_sum(I_MEAS + 6, I_MEAS + 7, 512);
    float* b_out = bias_sum(I_OUT_BI, I_OUT_BH, 512);

    // ---- activation scratch --------------------------------------------------
    float* XPAD = (float*)alloc((size_t)N_NOTES * IN_PAD * 4);
    float* XGF = (float*)alloc((size_t)N_NOTES * 1024 * 4);
    float* XGB = (float*)alloc((size_t)N_NOTES * 1024 * 4);
    float* HA  = (float*)alloc((size_t)N_NOTES * 512 * 4);
    float* HB  = (float*)alloc((size_t)N_NOTES * 512 * 4);   // final h_note
    float* ATT = (float*)alloc((size_t)N_NOTES * 512 * 4);
    float* BN  = (float*)alloc((size_t)N_BEATS * 512 * 4);
    float* HBA = (float*)alloc((size_t)N_BEATS * 512 * 4);
    float* HBB = (float*)alloc((size_t)N_BEATS * 512 * 4);   // final h_beat
    float* MN  = (float*)alloc((size_t)N_MEASQ * 512 * 4);
    float* HM  = (float*)alloc((size_t)N_MEASQ * 256 * 4);   // final h_meas
    int*   BM  = (int*)alloc((size_t)N_BEATS * 4);
    float* VLAST = (float*)alloc(512);

    // ================== note-level BiLSTM (2 layers, H=256) ==================
    k_padx<<<dim3(N_NOTES * IN_PAD / 256), dim3(256), 0, stream>>>(x, XPAD, N_NOTES * IN_PAD);
    gemm(XGF, 1024, XPAD, IN_PAD, nullptr, pWi_n0f, b_n0f, N_NOTES, IN_PAD, 1024, 0);
    gemm(XGB, 1024, XPAD, IN_PAD, nullptr, pWi_n0b, b_n0b, N_NOTES, IN_PAD, 1024, 0);
    k_lstm_scan<<<dim3(2), dim3(256), 0, stream>>>(XGF, XGB, pWh_n0f, pWh_n0b,
        HA, 512, nullptr, nullptr, nullptr, N_NOTES, 256);
    gemm(XGF, 1024, HA, 512, nullptr, pWi_n1f, b_n1f, N_NOTES, 512, 1024, 0);
    gemm(XGB, 1024, HA, 512, nullptr, pWi_n1b, b_n1b, N_NOTES, 512, 1024, 0);
    k_lstm_scan<<<dim3(2), dim3(256), 0, stream>>>(XGF, XGB, pWh_n1f, pWh_n1b,
        HB, 512, nullptr, nullptr, nullptr, N_NOTES, 256);

    // ================== notes -> beats attention pooling =====================
    gemm(ATT, 512, HB, 512, nullptr, pAttB, (const float*)d_in[I_BATT_B],
         N_NOTES, 512, 512, 0);
    k_segatt<<<dim3(N_BEATS), dim3(256), 0, stream>>>(HB, ATT, beatnum, N_NOTES, 512, BN);

    // ================== beat-level BiLSTM (2 layers, H=256) ==================
    gemm(XGF, 1024, BN, 512, nullptr, pWi_b0f, b_b0f, N_BEATS, 512, 1024, 0);
    gemm(XGB, 1024, BN, 512, nullptr, pWi_b0b, b_b0b, N_BEATS, 512, 1024, 0);
    k_lstm_scan<<<dim3(2), dim3(256), 0, stream>>>(XGF, XGB, pWh_b0f, pWh_b0b,
        HBA, 512, nullptr, nullptr, nullptr, N_BEATS, 256);
    gemm(XGF, 1024, HBA, 512, nullptr, pWi_b1f, b_b1f, N_BEATS, 512, 1024, 0);
    gemm(XGB, 1024, HBA, 512, nullptr, pWi_b1b, b_b1b, N_BEATS, 512, 1024, 0);
    k_lstm_scan<<<dim3(2), dim3(256), 0, stream>>>(XGF, XGB, pWh_b1f, pWh_b1b,
        HBB, 512, nullptr, nullptr, nullptr, N_BEATS, 256);

    // ================== beats -> measures attention pooling ==================
    k_first_measure<<<dim3((N_NOTES + 255) / 256), dim3(256), 0, stream>>>(
        beatnum, measnum, N_NOTES, BM);
    gemm(ATT, 512, HBB, 512, nullptr, pAttM, (const float*)d_in[I_MATT_B],
         N_BEATS, 512, 512, 0);
    k_segatt<<<dim3(N_MEASQ), dim3(256), 0, stream>>>(HBB, ATT, BM, N_BEATS, 512, MN);

    // ================== measure-level BiLSTM (1 layer, H=128) ================
    gemm(XGF, 512, MN, 512, nullptr, pWi_mf, b_mf, N_MEASQ, 512, 512, 0);
    gemm(XGB, 512, MN, 512, nullptr, pWi_mb, b_mb, N_MEASQ, 512, 512, 0);
    k_lstm_scan<<<dim3(2), dim3(256), 0, stream>>>(XGF, XGB, pWh_mf, pWh_mb,
        HM, 256, nullptr, nullptr, nullptr, N_MEASQ, 128);

    // ============== out LSTM: xg = combined @ Wi.T + b, via 3 GEMMs ==========
    float* XGO = XGF;  // reuse (needs 8192*512)
    gemm(XGO, 512, HB, 512, nullptr, pWiO_a, b_out, N_NOTES, 512, 512, 0);
    gemm(XGO, 512, HBB, 512, beatnum, pWiO_b, nullptr, N_NOTES, 512, 512, 1);
    gemm(XGO, 512, HM, 256, measnum, pWiO_c, nullptr, N_NOTES, 256, 512, 1);
    k_lstm_scan<<<dim3(1), dim3(256), 0, stream>>>(XGO, XGO, pWh_o, pWh_o,
        nullptr, 0, VLAST, fh0, fc0, N_NOTES, 128);

    // ================== head =================================================
    k_head<<<dim3(1), dim3(64), 0, stream>>>(VLAST, years,
        (const float*)d_in[I_FC1W], (const float*)d_in[I_FC1B],
        (const float*)d_in[I_FC2W], (const float*)d_in[I_FC2B],
        (float*)d_out);
}
